// QLinear_57793079935328
// MI455X (gfx1250) — matrix operand order repaired and checked by reading, not compiled, not run on hardware
//
#include <hip/hip_runtime.h>
#include <climits>
#include <cstdint>

// ---------------------------------------------------------------------------
// QLinear: y = x @ dequant(Wq)^T + dequant(bq)
// dequant affine:  w = s*q + c,  s = (wmax-wmin)/(qmax-qmin), c = wmin - s*qmin
// =>  y[b,o] = s * sum_i x[b,i]*q[o,i] + c * rowsum_x[b] + bias[o]
// Hot loop: f16 WMMA GEMM (int8-valued weights are EXACT in f16).
//
// Fast path (ws large enough): x and Wq are converted once to f16 in ws
// (fused into the reduction kernels), and the GEMM stages tiles with
// global_load_async_to_lds_b128 (ASYNCcnt) into triple-buffered LDS.
// Fallback path: convert during register staging (round-1 kernel).
// ---------------------------------------------------------------------------

typedef __attribute__((ext_vector_type(16))) _Float16 v16h;
typedef __attribute__((ext_vector_type(8)))  _Float16 v8h;
typedef __attribute__((ext_vector_type(4)))  _Float16 h4;
typedef __attribute__((ext_vector_type(8)))  float    v8f;
typedef __attribute__((ext_vector_type(4)))  float    f32x4;
typedef __attribute__((ext_vector_type(4)))  int      i32x4;

#define IN_F  4096
#define OUT_F 4096
#define BATCH 4096

#define BM 128
#define BN 128
#define BK 32
#define LDS_STRIDE 40   // 32 + 8 halfs pad; 80 bytes per row (16B-aligned rows)
#define KTILES (IN_F / BK)

// ws layout: [0..1] int q_min/q_max | +64B: rowsum[BATCH] f32 | biasf[OUT_F] f32
// fast path adds at byte 65536: xh[BATCH*IN_F] f16, wh[OUT_F*IN_F] f16
#define WS_F16_OFF   65536ull
#define WS_XH_BYTES  ((size_t)BATCH * IN_F * 2)
#define WS_WH_BYTES  ((size_t)OUT_F * IN_F * 2)
#define WS_REQUIRED  (WS_F16_OFF + WS_XH_BYTES + WS_WH_BYTES)

// ---------------------------------------------------------------------------
// Kernel 1: init workspace min/max slots (ws is poisoned before timing)
// ---------------------------------------------------------------------------
__global__ void qlin_init_kernel(int* wsI) {
    if (threadIdx.x == 0) {
        wsI[0] = INT_MAX;  // weight q_min
        wsI[1] = INT_MIN;  // weight q_max
    }
}

// ---------------------------------------------------------------------------
// Kernel 2: weight int32 min/max reduction (16M elements); optionally also
// converts weights to f16 (exact for int8-valued data).
// ---------------------------------------------------------------------------
__global__ void qlin_wminmax_kernel(const int* __restrict__ wq, int* wsI,
                                    _Float16* __restrict__ wh) {
    __shared__ int smn[256];
    __shared__ int smx[256];
    const int t = threadIdx.x;
    int lmn = INT_MAX, lmx = INT_MIN;
    const i32x4* w4 = (const i32x4*)wq;
    const size_t total4 = (size_t)OUT_F * IN_F / 4;          // 4,194,304
    size_t idx = (size_t)blockIdx.x * 256 + t;
    const size_t stride = (size_t)gridDim.x * 256;
    for (; idx < total4; idx += stride) {
        i32x4 v = w4[idx];
        int cmn = min(min(v[0], v[1]), min(v[2], v[3]));
        int cmx = max(max(v[0], v[1]), max(v[2], v[3]));
        lmn = min(lmn, cmn);
        lmx = max(lmx, cmx);
        if (wh) {
            h4 hv;
            #pragma unroll
            for (int j = 0; j < 4; ++j) hv[j] = (_Float16)(float)v[j];
            *(h4*)(wh + idx * 4) = hv;
        }
    }
    smn[t] = lmn; smx[t] = lmx;
    __syncthreads();
    for (int off = 128; off > 0; off >>= 1) {
        if (t < off) {
            smn[t] = min(smn[t], smn[t + off]);
            smx[t] = max(smx[t], smx[t + off]);
        }
        __syncthreads();
    }
    if (t == 0) {
        atomicMin(&wsI[0], smn[0]);
        atomicMax(&wsI[1], smx[0]);
    }
}

// ---------------------------------------------------------------------------
// Kernel 3: bias min/max + dequantize (single block, 256 threads, 4096 elems)
// ---------------------------------------------------------------------------
__global__ void qlin_bias_kernel(const int* __restrict__ bq,
                                 const float* __restrict__ bmin_p,
                                 const float* __restrict__ bmax_p,
                                 float* __restrict__ biasf) {
    __shared__ int smn[256];
    __shared__ int smx[256];
    const int t = threadIdx.x;
    int q[16];
    int lmn = INT_MAX, lmx = INT_MIN;
    #pragma unroll
    for (int i = 0; i < 16; ++i) {
        q[i] = bq[t + i * 256];
        lmn = min(lmn, q[i]);
        lmx = max(lmx, q[i]);
    }
    smn[t] = lmn; smx[t] = lmx;
    __syncthreads();
    for (int off = 128; off > 0; off >>= 1) {
        if (t < off) {
            smn[t] = min(smn[t], smn[t + off]);
            smx[t] = max(smx[t], smx[t + off]);
        }
        __syncthreads();
    }
    const int qmn = smn[0], qmx = smx[0];
    const float bmin = *bmin_p, bmax = *bmax_p;
    const float s = (bmax - bmin) / (float)(qmx - qmn);
    const float c = bmin - s * (float)qmn;
    #pragma unroll
    for (int i = 0; i < 16; ++i)
        biasf[t + i * 256] = s * (float)q[i] + c;
}

// ---------------------------------------------------------------------------
// Kernel 4: row sums of x (one block per row); optionally also converts x→f16
// ---------------------------------------------------------------------------
__global__ void qlin_rowsum_kernel(const float* __restrict__ x,
                                   float* __restrict__ rs,
                                   _Float16* __restrict__ xh) {
    __shared__ float red[256];
    const int row = blockIdx.x;
    const int t = threadIdx.x;
    const f32x4* xr = (const f32x4*)(x + (size_t)row * IN_F);
    float s = 0.f;
    #pragma unroll
    for (int i = 0; i < 4; ++i) {
        const int c = t + i * 256;
        f32x4 v = xr[c];
        s += v[0] + v[1] + v[2] + v[3];
        if (xh) {
            h4 hv;
            #pragma unroll
            for (int j = 0; j < 4; ++j) hv[j] = (_Float16)v[j];
            *(h4*)(xh + (size_t)row * IN_F + c * 4) = hv;
        }
    }
    red[t] = s;
    __syncthreads();
    for (int off = 128; off > 0; off >>= 1) {
        if (t < off) red[t] += red[t + off];
        __syncthreads();
    }
    if (t == 0) rs[row] = red[0];
}

// ---------------------------------------------------------------------------
// Fragment loads from LDS (wave32 WMMA layouts, ISA 7.12.2)
// ---------------------------------------------------------------------------
__device__ __forceinline__ v16h load_a_frag(const _Float16* Abuf, int row0, int lane) {
    const int r  = row0 + (lane & 15);
    const int kb = (lane >> 4) * 8;
    const _Float16* p = Abuf + r * LDS_STRIDE + kb;
    v8h lo = *(const v8h*)(p);
    v8h hi = *(const v8h*)(p + 16);
    return __builtin_shufflevector(lo, hi, 0,1,2,3,4,5,6,7,8,9,10,11,12,13,14,15);
}

__device__ __forceinline__ v16h load_b_frag(const _Float16* Bbuf, int col0, int lane) {
    const int r  = col0 + (lane & 15);
    const int kb = (lane >> 4) * 8;
    const _Float16* p = Bbuf + r * LDS_STRIDE + kb;
    v8h lo = *(const v8h*)(p);
    v8h hi = *(const v8h*)(p + 16);
    return __builtin_shufflevector(lo, hi, 0,1,2,3,4,5,6,7,8,9,10,11,12,13,14,15);
}

// Common epilogue: y = s*acc + c0*rowsum[row] + bias[col]
__device__ __forceinline__ void qlin_epilogue(
        v8f (&acc)[4][2], const int* wsI, const float* wmin_p, const float* wmax_p,
        const float* rowsum, const float* biasf, float* out,
        int rowBase, int colBase, int mBase, int nBase, int lane) {
    const int   qmn  = wsI[0];
    const int   qmx  = wsI[1];
    const float wmin = *wmin_p;
    const float wmax = *wmax_p;
    const float s  = (wmax - wmin) / (float)(qmx - qmn);
    const float c0 = wmin - s * (float)qmn;
    const int hi = lane >> 4;
    const int ln = lane & 15;
    #pragma unroll
    for (int mt = 0; mt < 4; ++mt) {
        #pragma unroll
        for (int nt = 0; nt < 2; ++nt) {
            const int col = colBase + nBase + nt * 16 + ln;
            const float bv = biasf[col];
            #pragma unroll
            for (int j = 0; j < 8; ++j) {
                const int row = rowBase + mBase + mt * 16 + j + 8 * hi;
                out[(size_t)row * OUT_F + col] =
                    s * acc[mt][nt][j] + c0 * rowsum[row] + bv;
            }
        }
    }
}

// ---------------------------------------------------------------------------
// Kernel 5a (fast path): WMMA GEMM on pre-converted f16, tiles staged with
// global_load_async_to_lds_b128 into triple-buffered LDS (ASYNCcnt tracked).
// Grid (32,32); block 256 = 8 waves in 2(M) x 4(N); wave tile 64x32.
// Per wave per K-step: 4 async-b128 issued; s_wait_asynccnt 4 retires the
// previous buffer's 4 (async loads complete in order); barrier; compute.
// Triple buffering makes write(kt+1 buf) vs read(kt-2 buf) barrier-separated.
// ---------------------------------------------------------------------------
__global__ __launch_bounds__(256)
void qlin_gemm_async_kernel(const _Float16* __restrict__ xh,
                            const _Float16* __restrict__ wh,
                            const int*   __restrict__ wsI,
                            const float* __restrict__ wmin_p,
                            const float* __restrict__ wmax_p,
                            const float* __restrict__ rowsum,
                            const float* __restrict__ biasf,
                            float* __restrict__ out) {
    __shared__ _Float16 As[3][BM * LDS_STRIDE];
    __shared__ _Float16 Bs[3][BN * LDS_STRIDE];

    const int tid    = threadIdx.x;
    const int lane   = tid & 31;
    const int waveId = tid >> 5;
    const int mBase  = (waveId >> 2) * 64;
    const int nBase  = (waveId & 3) * 32;

    const int rowBase = blockIdx.y * BM;
    const int colBase = blockIdx.x * BN;

    const uint64_t xh_base = (uint64_t)(uintptr_t)xh;
    const uint64_t wh_base = (uint64_t)(uintptr_t)wh;
    const unsigned ldsA0 = (unsigned)(uintptr_t)&As[0][0];
    const unsigned ldsB0 = (unsigned)(uintptr_t)&Bs[0][0];
    const unsigned BUFB  = (unsigned)(BM * LDS_STRIDE * sizeof(_Float16));

    // Per-thread chunk decomposition: 128 rows x 32 halfs = 512 x 16B chunks.
    // Thread handles chunks {tid, tid+256}: r = chunk/4, c8 = (chunk%4)*8.
    const int ch0_r  = tid >> 2;
    const int ch0_c8 = (tid & 3) * 8;
    const int ch1_r  = (tid + 256) >> 2;
    const int ch1_c8 = ((tid + 256) & 3) * 8;

    auto issue_tile = [&](int kk, int buf) {
        const unsigned la0 = ldsA0 + (unsigned)buf * BUFB +
                             (unsigned)((ch0_r * LDS_STRIDE + ch0_c8) * 2);
        const unsigned la1 = ldsA0 + (unsigned)buf * BUFB +
                             (unsigned)((ch1_r * LDS_STRIDE + ch1_c8) * 2);
        const unsigned lb0 = ldsB0 + (unsigned)buf * BUFB +
                             (unsigned)((ch0_r * LDS_STRIDE + ch0_c8) * 2);
        const unsigned lb1 = ldsB0 + (unsigned)buf * BUFB +
                             (unsigned)((ch1_r * LDS_STRIDE + ch1_c8) * 2);
        const unsigned ga0 = (unsigned)(((size_t)(rowBase + ch0_r) * IN_F + kk + ch0_c8) * 2);
        const unsigned ga1 = (unsigned)(((size_t)(rowBase + ch1_r) * IN_F + kk + ch1_c8) * 2);
        const unsigned gb0 = (unsigned)(((size_t)(colBase + ch0_r) * IN_F + kk + ch0_c8) * 2);
        const unsigned gb1 = (unsigned)(((size_t)(colBase + ch1_r) * IN_F + kk + ch1_c8) * 2);
        asm volatile("global_load_async_to_lds_b128 %0, %1, %2"
                     :: "v"(la0), "v"(ga0), "s"(xh_base) : "memory");
        asm volatile("global_load_async_to_lds_b128 %0, %1, %2"
                     :: "v"(la1), "v"(ga1), "s"(xh_base) : "memory");
        asm volatile("global_load_async_to_lds_b128 %0, %1, %2"
                     :: "v"(lb0), "v"(gb0), "s"(wh_base) : "memory");
        asm volatile("global_load_async_to_lds_b128 %0, %1, %2"
                     :: "v"(lb1), "v"(gb1), "s"(wh_base) : "memory");
    };

    v8f acc[4][2];
    #pragma unroll
    for (int mt = 0; mt < 4; ++mt)
        #pragma unroll
        for (int nt = 0; nt < 2; ++nt)
            #pragma unroll
            for (int j = 0; j < 8; ++j) acc[mt][nt][j] = 0.f;

    issue_tile(0, 0);

    for (int kt = 0; kt < KTILES; ++kt) {
        const int cur = kt % 3;
        if (kt + 1 < KTILES) {
            issue_tile((kt + 1) * BK, (kt + 1) % 3);
            asm volatile("s_wait_asynccnt 0x4" ::: "memory");  // retire cur's 4
        } else {
            asm volatile("s_wait_asynccnt 0x0" ::: "memory");
        }
        __syncthreads();

        v16h af[4], bf[2];
        #pragma unroll
        for (int mt = 0; mt < 4; ++mt)
            af[mt] = load_a_frag(As[cur], mBase + mt * 16, lane);
        #pragma unroll
        for (int nt = 0; nt < 2; ++nt)
            bf[nt] = load_b_frag(Bs[cur], nBase + nt * 16, lane);

        #pragma unroll
        for (int mt = 0; mt < 4; ++mt)
            #pragma unroll
            for (int nt = 0; nt < 2; ++nt)
                acc[mt][nt] = __builtin_amdgcn_wmma_f32_16x16x32_f16(
                    false, af[mt], false, bf[nt],
                    (short)0, acc[mt][nt], false, false);
    }

    qlin_epilogue(acc, wsI, wmin_p, wmax_p, rowsum, biasf, out,
                  rowBase, colBase, mBase, nBase, lane);
}

// ---------------------------------------------------------------------------
// Kernel 5b (fallback): convert during register staging (no f16 workspace).
// Double-buffered LDS, single barrier per K-step.
// ---------------------------------------------------------------------------
__global__ __launch_bounds__(256)
void qlin_gemm_kernel(const float* __restrict__ x,
                      const int*   __restrict__ wq,
                      const int*   __restrict__ wsI,
                      const float* __restrict__ wmin_p,
                      const float* __restrict__ wmax_p,
                      const float* __restrict__ rowsum,
                      const float* __restrict__ biasf,
                      float* __restrict__ out) {
    __shared__ _Float16 As[2][BM * LDS_STRIDE];
    __shared__ _Float16 Bs[2][BN * LDS_STRIDE];

    const int tid    = threadIdx.x;
    const int lane   = tid & 31;
    const int waveId = tid >> 5;
    const int mBase  = (waveId >> 2) * 64;
    const int nBase  = (waveId & 3) * 32;

    const int rowBase = blockIdx.y * BM;
    const int colBase = blockIdx.x * BN;

    v8f acc[4][2];
    #pragma unroll
    for (int mt = 0; mt < 4; ++mt)
        #pragma unroll
        for (int nt = 0; nt < 2; ++nt)
            #pragma unroll
            for (int j = 0; j < 8; ++j) acc[mt][nt][j] = 0.f;

    f32x4 xa[4];
    i32x4 wa[4];

    auto stage_load = [&](int kk) {
        #pragma unroll
        for (int c = 0; c < 4; ++c) {
            const int chunk = tid + c * 256;
            const int r  = chunk >> 3;
            const int c4 = (chunk & 7) * 4;
            xa[c] = *(const f32x4*)(x  + (size_t)(rowBase + r) * IN_F + kk + c4);
            wa[c] = *(const i32x4*)(wq + (size_t)(colBase + r) * IN_F + kk + c4);
        }
    };
    auto stage_store = [&](int buf) {
        #pragma unroll
        for (int c = 0; c < 4; ++c) {
            const int chunk = tid + c * 256;
            const int r  = chunk >> 3;
            const int c4 = (chunk & 7) * 4;
            h4 ha, hb;
            #pragma unroll
            for (int j = 0; j < 4; ++j) {
                ha[j] = (_Float16)xa[c][j];
                hb[j] = (_Float16)(float)wa[c][j];
            }
            *(h4*)(&As[buf][r * LDS_STRIDE + c4]) = ha;
            *(h4*)(&Bs[buf][r * LDS_STRIDE + c4]) = hb;
        }
    };

    stage_load(0);

    for (int kt = 0; kt < KTILES; ++kt) {
        const int buf = kt & 1;
        stage_store(buf);
        __syncthreads();
        if (kt + 1 < KTILES) stage_load((kt + 1) * BK);

        v16h af[4], bf[2];
        #pragma unroll
        for (int mt = 0; mt < 4; ++mt)
            af[mt] = load_a_frag(As[buf], mBase + mt * 16, lane);
        #pragma unroll
        for (int nt = 0; nt < 2; ++nt)
            bf[nt] = load_b_frag(Bs[buf], nBase + nt * 16, lane);

        #pragma unroll
        for (int mt = 0; mt < 4; ++mt)
            #pragma unroll
            for (int nt = 0; nt < 2; ++nt)
                acc[mt][nt] = __builtin_amdgcn_wmma_f32_16x16x32_f16(
                    false, af[mt], false, bf[nt],
                    (short)0, acc[mt][nt], false, false);
    }

    qlin_epilogue(acc, wsI, wmin_p, wmax_p, rowsum, biasf, out,
                  rowBase, colBase, mBase, nBase, lane);
}

// ---------------------------------------------------------------------------
extern "C" void kernel_launch(void* const* d_in, const int* in_sizes, int n_in,
                              void* d_out, int out_size, void* d_ws, size_t ws_size,
                              hipStream_t stream) {
    const float* x    = (const float*)d_in[0];
    const int*   wq   = (const int*)d_in[1];
    const int*   bq   = (const int*)d_in[2];
    const float* wmin = (const float*)d_in[3];
    const float* wmax = (const float*)d_in[4];
    const float* bmin = (const float*)d_in[5];
    const float* bmax = (const float*)d_in[6];
    float* out = (float*)d_out;

    int*   wsI    = (int*)d_ws;
    float* wsF    = (float*)d_ws;
    float* rowsum = wsF + 16;
    float* biasf  = wsF + 16 + BATCH;

    const bool fast = (ws_size >= WS_REQUIRED);
    _Float16* xh = fast ? (_Float16*)((char*)d_ws + WS_F16_OFF) : nullptr;
    _Float16* wh = fast ? (_Float16*)((char*)d_ws + WS_F16_OFF + WS_XH_BYTES) : nullptr;

    qlin_init_kernel<<<1, 64, 0, stream>>>(wsI);
    qlin_wminmax_kernel<<<256, 256, 0, stream>>>(wq, wsI, wh);
    qlin_bias_kernel<<<1, 256, 0, stream>>>(bq, bmin, bmax, biasf);
    qlin_rowsum_kernel<<<BATCH, 256, 0, stream>>>(x, rowsum, xh);

    dim3 grid(OUT_F / BN, BATCH / BM);
    if (fast) {
        qlin_gemm_async_kernel<<<grid, 256, 0, stream>>>(xh, wh, wsI, wmin, wmax,
                                                         rowsum, biasf, out);
    } else {
        qlin_gemm_kernel<<<grid, 256, 0, stream>>>(x, wq, wsI, wmin, wmax,
                                                   rowsum, biasf, out);
    }
}
